// TCN_81260781240587
// MI455X (gfx1250) — matrix operand order repaired and checked by reading, not compiled, not run on hardware
//
#include <hip/hip_runtime.h>
#include <math.h>

// ---------------------------------------------------------------------------
// CDNA5 / gfx1250 TCN implementation.
// Convs are im2col GEMMs on v_wmma_f32_16x16x32_bf16 (wave32), f32 accum,
// BN folded to per-channel alpha/beta, residual+ReLU fused in the epilogue.
// Conv kernels are fully templated so the K-loop unrolls into a static WMMA
// stream and all index math is shift/mask only.
// ---------------------------------------------------------------------------

typedef __attribute__((ext_vector_type(16))) __bf16 v16bf;
typedef __attribute__((ext_vector_type(8)))  __bf16 v8bf;
typedef __attribute__((ext_vector_type(8)))  float  v8f;

static constexpr int Bc  = 32;
static constexpr int Tc  = 4096;
static constexpr int Cc  = 256;   // hidden channels
static constexpr int NCc = 5;     // classes

// ---------------------------------------------------------------------------
// Prep: fold BN into alpha (scale) / beta (bias). alpha at [0..255], beta at
// [256..511] of each 512-float buffer. Downsample gets alpha=1, beta=bias.
// ---------------------------------------------------------------------------
__global__ void tcn_prep_scales(
    const float* c0b1, const float* c0g1, const float* c0bt1, const float* c0m1, const float* c0v1,
    const float* c0b2, const float* c0g2, const float* c0bt2, const float* c0m2, const float* c0v2,
    const float* c1b1, const float* c1g1, const float* c1bt1, const float* c1m1, const float* c1v1,
    const float* c1b2, const float* c1g2, const float* c1bt2, const float* c1m2, const float* c1v2,
    const float* c0db,
    float* s0a, float* s0b, float* s1a, float* s1b, float* sds)
{
    int c = threadIdx.x;
    if (c >= Cc) return;
    auto fold = [&](const float* bb, const float* g, const float* bt,
                    const float* m, const float* v, float* o) {
        float a = g[c] * rsqrtf(v[c] + 1e-5f);
        o[c]       = a;
        o[Cc + c]  = (bb[c] - m[c]) * a + bt[c];
    };
    fold(c0b1, c0g1, c0bt1, c0m1, c0v1, s0a);
    fold(c0b2, c0g2, c0bt2, c0m2, c0v2, s0b);
    fold(c1b1, c1g1, c1bt1, c1m1, c1v1, s1a);
    fold(c1b2, c1g2, c1bt2, c1m2, c1v2, s1b);
    sds[c]      = 1.0f;
    sds[Cc + c] = c0db[c];
}

// ---------------------------------------------------------------------------
// Weight repack: f32 (CO, CI, taps) -> bf16 WMMA B-fragment stream.
// Layout: [cot][kc][lane][16 bf16 contiguous].  Lane n = co%16; lane half
// selects K-slice; Kidx = kc*32 + (lane>>4)*16 + j ; K-dim = (tap, ci) with
// ci padded to CIP (zeros for ci >= CI).
// ---------------------------------------------------------------------------
template <int CI, int TAPS, int CIP>
__global__ void tcn_repack_w(const float* __restrict__ w, __bf16* __restrict__ out)
{
    constexpr int nK    = (TAPS * CIP) >> 5;
    constexpr int total = 16 * nK * 512;
    for (int idx = blockIdx.x * 256 + threadIdx.x; idx < total;
         idx += gridDim.x * 256) {
        int j    = idx & 15;
        int lane = (idx >> 4) & 31;
        int kc   = (idx >> 9) % nK;
        int cot  = (idx >> 9) / nK;
        int co   = cot * 16 + (lane & 15);
        int Kidx = kc * 32 + ((lane >> 4) << 4) + j;
        int kk   = Kidx / CIP;
        int ci   = Kidx % CIP;
        float v  = (ci < CI) ? w[((size_t)co * CI + ci) * TAPS + kk] : 0.0f;
        out[idx] = (__bf16)v;
    }
}

// Convert input x (B,T,34) f32 -> (B,T,64) bf16, zero-padded channels.
__global__ void tcn_cvt_x(const float* __restrict__ x, __bf16* __restrict__ X0)
{
    size_t idx = (size_t)blockIdx.x * 256 + threadIdx.x;   // over B*T*64
    size_t bt  = idx >> 6;
    int    c   = (int)(idx & 63);
    X0[idx] = (c < 34) ? (__bf16)x[bt * 34 + c] : (__bf16)0.0f;
}

// ---------------------------------------------------------------------------
// Causal dilated conv as WMMA GEMM (templated: all loops unroll statically).
//   grid.x = B * (T/16) ; block = 256 (8 wave32)
//   block: one batch b, 16 timesteps, ALL 256 output channels.
//   wave wv -> co tiles 2*wv and 2*wv+1 (16x32 output), sharing A fragments.
//   A (16 x TAPS*CIP) causal im2col tile staged once in LDS.
//   Epilogue: alpha*acc+beta, optional ReLU, optional residual (+ReLU), bf16.
// ---------------------------------------------------------------------------
template <int CIP, int TAPS, int DIL, bool RELU, bool HAS_RES>
__global__ __launch_bounds__(256) void tcn_conv_wmma(
    const __bf16* __restrict__ X, const __bf16* __restrict__ Wf,
    const float* __restrict__ scales,          // [0..255]=alpha [256..511]=beta
    const __bf16* __restrict__ Res,
    __bf16* __restrict__ Y)
{
    constexpr int KD   = TAPS * CIP;           // K-dim
    constexpr int nK   = KD >> 5;              // 32-deep chunks
    constexpr int CIP8 = CIP >> 3;
    constexpr int CHUNKS = 16 * TAPS * CIP8;   // 8-elem vectors to stage

    __shared__ __bf16 At[16][KD + 8];

    const int bid = blockIdx.x;
    const int tt  = bid & ((Tc / 16) - 1);
    const int b   = bid >> 8;                  // T/16 = 256
    const int t0  = tt * 16;

    // Stage causal im2col A tile (16 rows x TAPS*CIP) into LDS.
    #pragma unroll
    for (int it = 0; it < (CHUNKS + 255) / 256; ++it) {
        int idx = it * 256 + threadIdx.x;
        if (CHUNKS % 256 == 0 || idx < CHUNKS) {
            int m   = idx / (TAPS * CIP8);
            int rem = idx - m * (TAPS * CIP8);
            int kk  = rem / CIP8;
            int c8  = rem - kk * CIP8;
            int ts  = t0 + m - (TAPS - 1 - kk) * DIL;   // causal left pad
            v8bf v = {};
            if (ts >= 0)
                v = *(const v8bf*)(X + ((size_t)(b * Tc + ts) * CIP + c8 * 8));
            *(v8bf*)(&At[m][kk * CIP + c8 * 8]) = v;
        }
    }
    __syncthreads();

    const int lane = threadIdx.x & 31;
    const int wv   = threadIdx.x >> 5;
    const int h    = lane >> 4;
    const int mn   = lane & 15;

    v8f acc0 = {};
    v8f acc1 = {};
    const __bf16* wb0 = Wf + (size_t)(2 * wv)     * nK * 512;
    const __bf16* wb1 = Wf + (size_t)(2 * wv + 1) * nK * 512;

    #pragma unroll
    for (int kc = 0; kc < nK; ++kc) {
        // A fragment (contiguous K order, k = 16*h + s, as in B repack): lane = row M, half = K-slice.
        const __bf16* ap = &At[mn][kc * 32 + h * 16];
        v8bf lo = *(const v8bf*)(ap);
        v8bf hi = *(const v8bf*)(ap + 8);
        v16bf a = __builtin_shufflevector(lo, hi,
                      0, 1, 2, 3, 4, 5, 6, 7, 8, 9, 10, 11, 12, 13, 14, 15);
        // B fragments: 16 contiguous bf16 per lane from repacked stream.
        v16bf bm0 = *(const v16bf*)(wb0 + ((size_t)kc * 32 + lane) * 16);
        v16bf bm1 = *(const v16bf*)(wb1 + ((size_t)kc * 32 + lane) * 16);
        if (kc + 2 < nK) {
            __builtin_prefetch(wb0 + ((size_t)(kc + 2) * 32 + lane) * 16, 0, 1);
            __builtin_prefetch(wb1 + ((size_t)(kc + 2) * 32 + lane) * 16, 0, 1);
        }
        acc0 = __builtin_amdgcn_wmma_f32_16x16x32_bf16(
                   false, a, false, bm0, (short)0, acc0, false, false);
        acc1 = __builtin_amdgcn_wmma_f32_16x16x32_bf16(
                   false, a, false, bm1, (short)0, acc1, false, false);
    }

    // Epilogue (C/D layout: col = lane&15, row = r + 8*(lane>=16)).
    const size_t base = (size_t)b * Tc;
    #pragma unroll
    for (int tile = 0; tile < 2; ++tile) {
        const int co   = (2 * wv + tile) * 16 + mn;
        const float al = scales[co];
        const float be = scales[Cc + co];
        const v8f&  acc = tile ? acc1 : acc0;
        #pragma unroll
        for (int r = 0; r < 8; ++r) {
            int t   = t0 + r + h * 8;
            float v = acc[r] * al + be;
            if (RELU) v = fmaxf(v, 0.0f);
            if (HAS_RES) {
                float rv = (float)Res[(base + t) * Cc + co];
                v = fmaxf(v + rv, 0.0f);
            }
            Y[(base + t) * Cc + co] = (__bf16)v;
        }
    }
}

// ---------------------------------------------------------------------------
// Head: attn = sigmoid(z.aw + ab); logits = attn*(z@lw) + lb; softmax;
// preds = argmax; alt = argmax excluding class 3.  One wave32 per (b,t).
// ---------------------------------------------------------------------------
__global__ __launch_bounds__(256) void tcn_head(
    const __bf16* __restrict__ Z, const float* __restrict__ aw,
    const float* __restrict__ ab, const float* __restrict__ lw,
    const float* __restrict__ lb, float* __restrict__ logits_out,
    float* __restrict__ probs, int* __restrict__ preds, int* __restrict__ alt)
{
    const int lane = threadIdx.x & 31;
    const int wv   = threadIdx.x >> 5;
    const size_t row = (size_t)blockIdx.x * 8 + wv;

    v8bf zv = *(const v8bf*)(Z + row * Cc + lane * 8);
    float pa = 0.0f, pl[NCc] = {0, 0, 0, 0, 0};
    #pragma unroll
    for (int j = 0; j < 8; ++j) {
        float z  = (float)zv[j];
        int   ci = lane * 8 + j;
        pa += z * aw[ci];
        #pragma unroll
        for (int c = 0; c < NCc; ++c) pl[c] += z * lw[ci * NCc + c];
    }
    #pragma unroll
    for (int off = 16; off > 0; off >>= 1) {
        pa += __shfl_down(pa, off, 32);
        #pragma unroll
        for (int c = 0; c < NCc; ++c) pl[c] += __shfl_down(pl[c], off, 32);
    }
    if (lane == 0) {
        float attn = 1.0f / (1.0f + expf(-(pa + ab[0])));
        float l[NCc], mx = -3.4e38f;
        #pragma unroll
        for (int c = 0; c < NCc; ++c) {
            l[c] = attn * pl[c] + lb[c];
            logits_out[row * NCc + c] = l[c];
            mx = fmaxf(mx, l[c]);
        }
        float s = 0.0f;
        #pragma unroll
        for (int c = 0; c < NCc; ++c) { l[c] = expf(l[c] - mx); s += l[c]; }
        float inv = 1.0f / s;
        int pmax = 0, amax = 0;
        float pb = -1.0f, abst = -1.0f;
        #pragma unroll
        for (int c = 0; c < NCc; ++c) {
            float p = l[c] * inv;
            probs[row * NCc + c] = p;
            if (p > pb) { pb = p; pmax = c; }
            if (c != 3 && p > abst) { abst = p; amax = c; }
        }
        preds[row] = pmax;
        alt[row]   = amax;
    }
}

// Window smoothing: 5-wide window of (pred != 3), majority -> replace NO_ACT.
__global__ void tcn_smooth(const int* __restrict__ preds,
                           const int* __restrict__ alt, int* __restrict__ proc)
{
    int idx = blockIdx.x * 256 + threadIdx.x;       // b*T + t
    int t = idx & (Tc - 1);
    int b = idx >> 12;
    int s = max(0, t - 2), e = min(Tc, t + 3);
    int wsum = 0;
    for (int u = s; u < e; ++u) wsum += (preds[b * Tc + u] != 3) ? 1 : 0;
    int p = preds[idx];
    proc[idx] = (p == 3 && 2 * wsum > (e - s)) ? alt[idx] : p;
}

// Sequential transition-matrix scan, one thread per batch row.
// TM = ones(5,5) with TM[1][2]=0, TM[2][0]=0.
__global__ void tcn_scan(const int* __restrict__ proc,
                         const float* __restrict__ probs,
                         float* __restrict__ outp)
{
    int b = threadIdx.x;
    if (b >= Bc) return;
    int prev = proc[b * Tc];
    outp[b * Tc] = (float)prev;
    for (int t = 1; t < Tc; ++t) {
        int curr = proc[b * Tc + t];
        bool invalid = (prev == 1 && curr == 2) || (prev == 2 && curr == 0);
        int nw = curr;
        if (invalid) {
            const float* p = probs + ((size_t)b * Tc + t) * NCc;
            float best = -1.0f, s = 0.0f;
            int bi = 0;
            for (int c = 0; c < NCc; ++c) {
                bool masked = (prev == 1 && c == 2) || (prev == 2 && c == 0);
                float vp = masked ? 0.0f : p[c];
                s += vp;
                if (vp > best) { best = vp; bi = c; }
            }
            nw = (s == 0.0f) ? prev : bi;
        }
        outp[b * Tc + t] = (float)nw;
        prev = nw;
    }
}

// ---------------------------------------------------------------------------
extern "C" void kernel_launch(void* const* d_in, const int* in_sizes, int n_in,
                              void* d_out, int out_size, void* d_ws, size_t ws_size,
                              hipStream_t stream)
{
    const float* x    = (const float*)d_in[0];
    const float* c0w1 = (const float*)d_in[1];
    const float* c0b1 = (const float*)d_in[2];
    const float* c0w2 = (const float*)d_in[3];
    const float* c0b2 = (const float*)d_in[4];
    const float* c0g1 = (const float*)d_in[5];
    const float* c0bt1= (const float*)d_in[6];
    const float* c0m1 = (const float*)d_in[7];
    const float* c0v1 = (const float*)d_in[8];
    const float* c0g2 = (const float*)d_in[9];
    const float* c0bt2= (const float*)d_in[10];
    const float* c0m2 = (const float*)d_in[11];
    const float* c0v2 = (const float*)d_in[12];
    const float* c1w1 = (const float*)d_in[13];
    const float* c1b1 = (const float*)d_in[14];
    const float* c1w2 = (const float*)d_in[15];
    const float* c1b2 = (const float*)d_in[16];
    const float* c1g1 = (const float*)d_in[17];
    const float* c1bt1= (const float*)d_in[18];
    const float* c1m1 = (const float*)d_in[19];
    const float* c1v1 = (const float*)d_in[20];
    const float* c1g2 = (const float*)d_in[21];
    const float* c1bt2= (const float*)d_in[22];
    const float* c1m2 = (const float*)d_in[23];
    const float* c1v2 = (const float*)d_in[24];
    const float* c0dw = (const float*)d_in[25];
    const float* c0db = (const float*)d_in[26];
    const float* aw   = (const float*)d_in[27];
    const float* ab   = (const float*)d_in[28];
    const float* lw   = (const float*)d_in[29];
    const float* lb   = (const float*)d_in[30];

    const size_t BT = (size_t)Bc * Tc;

    // Workspace carve (256B aligned).
    char* wp = (char*)d_ws;
    auto carve = [&](size_t bytes) {
        char* p = wp;
        wp += (bytes + 255) & ~(size_t)255;
        return p;
    };
    __bf16* X0  = (__bf16*)carve(BT * 64 * 2);       // padded bf16 input
    __bf16* Ha  = (__bf16*)carve(BT * Cc * 2);       // conv1 / conv3 out
    __bf16* Hb  = (__bf16*)carve(BT * Cc * 2);       // block0 out
    __bf16* Zb  = (__bf16*)carve(BT * Cc * 2);       // Rd residual, then block1 out
    float*  probs = (float*)carve(BT * NCc * 4);
    int*    preds = (int*)carve(BT * 4);
    int*    altb  = (int*)carve(BT * 4);
    int*    procb = (int*)carve(BT * 4);
    __bf16* Wf0a = (__bf16*)carve((size_t)16 * 6 * 512 * 2);   // 34->256 K3, CIP64
    __bf16* Wfds = (__bf16*)carve((size_t)16 * 2 * 512 * 2);   // 34->256 K1, CIP64
    __bf16* Wf0b = (__bf16*)carve((size_t)16 * 24 * 512 * 2);  // 256->256 K3
    __bf16* Wf1a = (__bf16*)carve((size_t)16 * 24 * 512 * 2);
    __bf16* Wf1b = (__bf16*)carve((size_t)16 * 24 * 512 * 2);
    float* s0a = (float*)carve(512 * 4);
    float* s0b = (float*)carve(512 * 4);
    float* s1a = (float*)carve(512 * 4);
    float* s1b = (float*)carve(512 * 4);
    float* sds = (float*)carve(512 * 4);

    // --- prep ---
    tcn_prep_scales<<<1, 256, 0, stream>>>(
        c0b1, c0g1, c0bt1, c0m1, c0v1,  c0b2, c0g2, c0bt2, c0m2, c0v2,
        c1b1, c1g1, c1bt1, c1m1, c1v1,  c1b2, c1g2, c1bt2, c1m2, c1v2,
        c0db, s0a, s0b, s1a, s1b, sds);
    tcn_repack_w<34, 3, 64><<<192, 256, 0, stream>>>(c0w1, Wf0a);
    tcn_repack_w<34, 1, 64><<<64, 256, 0, stream>>>(c0dw, Wfds);
    tcn_repack_w<256, 3, 256><<<768, 256, 0, stream>>>(c0w2, Wf0b);
    tcn_repack_w<256, 3, 256><<<768, 256, 0, stream>>>(c1w1, Wf1a);
    tcn_repack_w<256, 3, 256><<<768, 256, 0, stream>>>(c1w2, Wf1b);
    tcn_cvt_x<<<(int)(BT * 64 / 256), 256, 0, stream>>>(x, X0);

    // --- conv stack (grid: B * T/16 blocks; each block does all 256 co) ---
    const int cgrid = Bc * (Tc / 16);
    // Block0 conv1: 34->256, K3 d1, BN+ReLU
    tcn_conv_wmma<64, 3, 1, true, false>
        <<<cgrid, 256, 0, stream>>>(X0, Wf0a, s0a, nullptr, Ha);
    // Downsample residual: 34->256, 1x1, bias only
    tcn_conv_wmma<64, 1, 1, false, false>
        <<<cgrid, 256, 0, stream>>>(X0, Wfds, sds, nullptr, Zb);
    // Block0 conv2: BN+ReLU then +res, ReLU
    tcn_conv_wmma<256, 3, 1, true, true>
        <<<cgrid, 256, 0, stream>>>(Ha, Wf0b, s0b, Zb, Hb);
    // Block1 conv1: K3 d2, BN+ReLU
    tcn_conv_wmma<256, 3, 2, true, false>
        <<<cgrid, 256, 0, stream>>>(Hb, Wf1a, s1a, nullptr, Ha);
    // Block1 conv2: K3 d2, BN+ReLU then +identity res (Hb), ReLU
    tcn_conv_wmma<256, 3, 2, true, true>
        <<<cgrid, 256, 0, stream>>>(Ha, Wf1b, s1b, Hb, Zb);

    // --- head + post ---
    float* logits_out = (float*)d_out;
    float* proc_out   = (float*)d_out + BT * NCc;
    tcn_head<<<(int)(BT / 8), 256, 0, stream>>>(Zb, aw, ab, lw, lb,
                                                logits_out, probs, preds, altb);
    tcn_smooth<<<(int)(BT / 256), 256, 0, stream>>>(preds, altb, procb);
    tcn_scan<<<1, 32, 0, stream>>>(procb, probs, proc_out);
}
